// CBAMBlock_55980603736223
// MI455X (gfx1250) — matrix operand order repaired and checked by reading, not compiled, not run on hardware
//
#include <hip/hip_runtime.h>
#include <hip/hip_bf16.h>
#include <math.h>

// ---------------------------------------------------------------------------
// RetNet block for MI455X (gfx1250, wave32, WMMA).
// fc GEMM (128 x 49152) @ (49152 x 2048) streams 402MB of fc_w => HBM-bound
// (~17us @ 23.3 TB/s). A is pre-packed to f16 once so the hot loop is a pure
// fc_w stream + WMMA; A staging uses GLOBAL_LOAD_ASYNC_TO_LDS_B128.
// ---------------------------------------------------------------------------

typedef _Float16 v16h __attribute__((ext_vector_type(16)));
typedef _Float16 v8h  __attribute__((ext_vector_type(8)));
typedef float    v8f  __attribute__((ext_vector_type(8)));
typedef int      v4i  __attribute__((ext_vector_type(4)));

union V16 { v16h v; v8h h[2]; };

#define MDIM   128          // batch
#define NDIM   2048         // 256*H
#define KTOT   49152        // 6144*H
#define KSPLIT 8
#define KPER   (KTOT / KSPLIT)   // 6144
#define TILE_N 64
#define LDP    40           // LDS row pitch in halves (80B: 16B aligned, conflict-free)

#define AS1 __attribute__((address_space(1)))
#define AS3 __attribute__((address_space(3)))

#if defined(__has_builtin)
#  if __has_builtin(__builtin_amdgcn_global_load_async_to_lds_b128) && \
      __has_builtin(__builtin_amdgcn_s_wait_asynccnt)
#    define USE_ASYNC 1
#  else
#    define USE_ASYNC 0
#  endif
#else
#  define USE_ASYNC 0
#endif

// ---------------------------------------------------------------------------
// Kernel 0: pack A once: Apk[m][k] = (f16) x[m, k&7, k>>3].
// Thread = (m, d): 8 coalesced f32 reads, one contiguous b128 f16 write.
// ---------------------------------------------------------------------------
__global__ __launch_bounds__(256) void pack_a_kernel(
    const float* __restrict__ x, _Float16* __restrict__ Apk)
{
    int idx = blockIdx.x * 256 + threadIdx.x;    // 0 .. 128*6144-1
    int m = idx / 6144;
    int d = idx - m * 6144;
    _Float16 tmp[8];
    #pragma unroll
    for (int hh = 0; hh < 8; ++hh)
        tmp[hh] = (_Float16)x[(size_t)m * KTOT + (size_t)hh * 6144 + d];
    *reinterpret_cast<v8h*>(Apk + (size_t)m * KTOT + (size_t)d * 8) =
        *reinterpret_cast<v8h*>(tmp);
}

// ---------------------------------------------------------------------------
// Kernel 1: fc GEMM partials. Grid (NDIM/TILE_N, KSPLIT) x 256 threads.
// WG computes a 128x64 f32 partial tile over its K range using WMMA f16.
// ---------------------------------------------------------------------------
__global__ __launch_bounds__(256) void fc_gemm_kernel(
    const _Float16* __restrict__ Apk,  // (128, 49152) f16, k-contiguous
    const float* __restrict__ fc_w,    // (49152, 2048)
    float* __restrict__ partial)       // (KSPLIT, 128, 2048)
{
    __shared__ _Float16 As[MDIM * LDP];    // A tile 128x32 (f16)
    __shared__ _Float16 Bs[TILE_N * LDP];  // B tile stored [n][k] 64x32 (f16)

    const int t    = threadIdx.x;
    const int wave = t >> 5;
    const int lane = t & 31;
    const int n0   = blockIdx.x * TILE_N;
    const int kseg = blockIdx.y * KPER;

    // A-stage slots for this thread (2 x 16B): c = m*4 + quarter
    const int ac0_m = t >> 2,          ac0_o = (t & 3) * 8;
    const int ac1_m = (t + 256) >> 2,  ac1_o = ((t + 256) & 3) * 8;

    v8f acc[4] = {};

    for (int kc = 0; kc < KPER; kc += 32) {
        const int kb = kseg + kc;

        // ---- stage A (pure f16 byte copy; async-to-LDS, ASYNCcnt tracked) ----
#if USE_ASYNC
        __builtin_amdgcn_global_load_async_to_lds_b128(
            (AS1 v4i*)(Apk + (size_t)ac0_m * KTOT + kb + ac0_o),
            (AS3 v4i*)(&As[ac0_m * LDP + ac0_o]), 0, 0);
        __builtin_amdgcn_global_load_async_to_lds_b128(
            (AS1 v4i*)(Apk + (size_t)ac1_m * KTOT + kb + ac1_o),
            (AS3 v4i*)(&As[ac1_m * LDP + ac1_o]), 0, 0);
#else
        *reinterpret_cast<v8h*>(&As[ac0_m * LDP + ac0_o]) =
            *reinterpret_cast<const v8h*>(Apk + (size_t)ac0_m * KTOT + kb + ac0_o);
        *reinterpret_cast<v8h*>(&As[ac1_m * LDP + ac1_o]) =
            *reinterpret_cast<const v8h*>(Apk + (size_t)ac1_m * KTOT + kb + ac1_o);
#endif

        // ---- stage B: 2 x float4 loads, convert, transpose-scatter [n][k] ----
        #pragma unroll
        for (int i = 0; i < 2; ++i) {
            int f  = t + 256 * i;       // 0..511
            int kk = f >> 4;            // 0..31
            int nn = (f & 15) * 4;      // 0..60
            const float4 v = *reinterpret_cast<const float4*>(
                &fc_w[(size_t)(kb + kk) * NDIM + n0 + nn]);
            Bs[(nn + 0) * LDP + kk] = (_Float16)v.x;
            Bs[(nn + 1) * LDP + kk] = (_Float16)v.y;
            Bs[(nn + 2) * LDP + kk] = (_Float16)v.z;
            Bs[(nn + 3) * LDP + kk] = (_Float16)v.w;
        }
        // prefetch next fc_w chunk (global_prefetch_b8)
        if (kc + 32 < KPER) {
            __builtin_prefetch(
                &fc_w[(size_t)(kb + 32 + (t >> 4)) * NDIM + n0 + (t & 15) * 4], 0, 0);
        }
#if USE_ASYNC
        __builtin_amdgcn_s_wait_asynccnt(0);
#endif
        __syncthreads();

        // ---- A fragment (16-bit A 16x32 layout): lane<16 -> K 0..7,16..23 ;
        //      lane>=16 -> K 8..15,24..31 ----
        const int mrow = wave * 16 + (lane & 15);
        const int akb  = (lane < 16) ? 0 : 8;
        V16 a;
        a.h[0] = *reinterpret_cast<const v8h*>(&As[mrow * LDP + akb]);
        a.h[1] = *reinterpret_cast<const v8h*>(&As[mrow * LDP + akb + 16]);

        // ---- B fragments: lane<16 -> K 0..7,16..23 ; lane>=16 -> K 8..15,24..31 ----
        const int bkb = (lane < 16) ? 0 : 8;
        #pragma unroll
        for (int nt = 0; nt < 4; ++nt) {
            const int ncol = nt * 16 + (lane & 15);
            V16 b;
            b.h[0] = *reinterpret_cast<const v8h*>(&Bs[ncol * LDP + bkb]);
            b.h[1] = *reinterpret_cast<const v8h*>(&Bs[ncol * LDP + bkb + 16]);
            acc[nt] = __builtin_amdgcn_wmma_f32_16x16x32_f16(
                false, a.v, false, b.v, (short)0, acc[nt], false, false);
        }
        __syncthreads();
    }

    // ---- epilogue: C/D layout: VGPR r, lane l -> M = r + 8*(l>>4), N = l&15
    const int mbase = wave * 16 + ((lane >> 4) << 3);
    const int ncolb = lane & 15;
    float* pp = partial + (size_t)blockIdx.y * MDIM * NDIM;
    #pragma unroll
    for (int nt = 0; nt < 4; ++nt) {
        #pragma unroll
        for (int r = 0; r < 8; ++r) {
            pp[(size_t)(mbase + r) * NDIM + (n0 + nt * 16 + ncolb)] = acc[nt][r];
        }
    }
}

// ---------------------------------------------------------------------------
// Kernel 2: reduce K-split partials + bias -> X (== h reshaped (B,L,H))
// ---------------------------------------------------------------------------
__global__ __launch_bounds__(256) void fc_reduce_kernel(
    const float* __restrict__ partial, const float* __restrict__ fc_b,
    float* __restrict__ X)
{
    int i = blockIdx.x * 256 + threadIdx.x;     // 0 .. 128*2048-1
    float s = fc_b[i & (NDIM - 1)];
    #pragma unroll
    for (int ks = 0; ks < KSPLIT; ++ks)
        s += partial[(size_t)ks * MDIM * NDIM + i];
    X[i] = s;
}

// ---------------------------------------------------------------------------
// Kernel 3: LayerNorm over H=8 -> Xn. One row per thread.
// ---------------------------------------------------------------------------
__global__ __launch_bounds__(256) void ln1_kernel(
    const float* __restrict__ X, const float* __restrict__ s,
    const float* __restrict__ b, float* __restrict__ Xn)
{
    int row = blockIdx.x * 256 + threadIdx.x;   // 0..32767
    const float* xr = X + (size_t)row * 8;
    float xv[8], m = 0.f;
    #pragma unroll
    for (int j = 0; j < 8; ++j) { xv[j] = xr[j]; m += xv[j]; }
    m *= 0.125f;
    float v = 0.f;
    #pragma unroll
    for (int j = 0; j < 8; ++j) { float d = xv[j] - m; v += d * d; }
    v *= 0.125f;
    float inv = rsqrtf(v + 1e-5f);
    float* o = Xn + (size_t)row * 8;
    #pragma unroll
    for (int j = 0; j < 8; ++j) o[j] = (xv[j] - m) * inv * s[j] + b[j];
}

// ---------------------------------------------------------------------------
// Kernel 4: retention head. Grid (B=128, HEADS=2) x 256 threads (1 query/thr).
// xpos rotation in registers; K,V staged in LDS; causal decay via running w.
// ---------------------------------------------------------------------------
__global__ __launch_bounds__(256) void retention_kernel(
    const float* __restrict__ Xn, const float* __restrict__ Wq,
    const float* __restrict__ Wk, const float* __restrict__ Wv,
    float* __restrict__ Yc)
{
    __shared__ float Ks[256][4];
    __shared__ float Vs[256][8];

    const int b = blockIdx.x, head = blockIdx.y;
    const int l = threadIdx.x;

    const float* xr = Xn + ((size_t)b * 256 + l) * 8;
    float xv[8];
    #pragma unroll
    for (int i = 0; i < 8; ++i) xv[i] = xr[i];

    const float* wq = Wq + head * 32;   // (8,4)
    const float* wk = Wk + head * 32;   // (8,4)
    const float* wv = Wv + head * 64;   // (8,8)

    float q[4] = {}, k[4] = {}, v[8] = {};
    #pragma unroll
    for (int i = 0; i < 8; ++i) {
        #pragma unroll
        for (int j = 0; j < 4; ++j) { q[j] += xv[i] * wq[i * 4 + j]; k[j] += xv[i] * wk[i * 4 + j]; }
        #pragma unroll
        for (int j = 0; j < 8; ++j) v[j] += xv[i] * wv[i * 8 + j];
    }

    // xpos: Dh=4, sv=[1.6/5.6, 3.6/5.6], inv_freq=[1, 0.01]
    const float pos = (float)l;
    const float e   = pos * (1.f / 512.f);
    const float sc0 = __expf(-1.2527630f * e);   // ln(1.6/5.6)
    const float sc1 = __expf(-0.4418328f * e);   // ln(3.6/5.6)
    const float c0 = cosf(pos),         s0 = sinf(pos);
    const float c1 = cosf(pos * 0.01f), s1 = sinf(pos * 0.01f);

    // Q: scale ; K: 1/scale
    {
        float a0 = c0 * sc0, b0 = s0 * sc0, a1 = c1 * sc1, b1 = s1 * sc1;
        float q0 = q[0] * a0 - q[1] * b0, q1 = q[1] * a0 + q[0] * b0;
        float q2 = q[2] * a1 - q[3] * b1, q3 = q[3] * a1 + q[2] * b1;
        q[0] = q0; q[1] = q1; q[2] = q2; q[3] = q3;
        float i0 = 1.f / sc0, i1 = 1.f / sc1;
        a0 = c0 * i0; b0 = s0 * i0; a1 = c1 * i1; b1 = s1 * i1;
        float k0 = k[0] * a0 - k[1] * b0, k1 = k[1] * a0 + k[0] * b0;
        float k2 = k[2] * a1 - k[3] * b1, k3 = k[3] * a1 + k[2] * b1;
        k[0] = k0; k[1] = k1; k[2] = k2; k[3] = k3;
    }

    #pragma unroll
    for (int j = 0; j < 4; ++j) Ks[l][j] = k[j];
    #pragma unroll
    for (int j = 0; j < 8; ++j) Vs[l][j] = v[j];
    __syncthreads();

    const float gamma = (head == 0) ? (1.f - 1.f / 32.f) : (1.f - 1.f / 512.f);
    float acc[8] = {};
    float w = 1.f;
    for (int m = l; m >= 0; --m) {
        float sc = (q[0] * Ks[m][0] + q[1] * Ks[m][1] +
                    q[2] * Ks[m][2] + q[3] * Ks[m][3]) * w;
        #pragma unroll
        for (int j = 0; j < 8; ++j) acc[j] += sc * Vs[m][j];
        w *= gamma;
    }

    float* yr = Yc + ((size_t)b * 256 + l) * 16 + head * 8;
    #pragma unroll
    for (int j = 0; j < 8; ++j) yr[j] = acc[j];
}

// ---------------------------------------------------------------------------
// Kernel 5: fused groupnorm + swish gate + Wo + residual + LN2 + FFN + residual
// One row (b,l) per thread; tiny weights staged in LDS.
// ---------------------------------------------------------------------------
__global__ __launch_bounds__(256) void post_kernel(
    const float* __restrict__ X,  const float* __restrict__ Xn,
    const float* __restrict__ Yc,
    const float* __restrict__ Wg, const float* __restrict__ Wo,
    const float* __restrict__ gn_w, const float* __restrict__ gn_b,
    const float* __restrict__ ln2_s, const float* __restrict__ ln2_b,
    const float* __restrict__ w1, const float* __restrict__ b1,
    const float* __restrict__ w2, const float* __restrict__ b2,
    float* __restrict__ Xout)
{
    __shared__ float sWg[128], sWo[128], sgw[16], sgb[16];
    __shared__ float sl2s[8], sl2b[8], sw1[64], sb1[8], sw2[64], sb2[8];
    const int t = threadIdx.x;
    if (t < 128) { sWg[t] = Wg[t]; sWo[t] = Wo[t]; }
    if (t < 64)  { sw1[t] = w1[t]; sw2[t] = w2[t]; }
    if (t < 16)  { sgw[t] = gn_w[t]; sgb[t] = gn_b[t]; }
    if (t < 8)   { sl2s[t] = ln2_s[t]; sl2b[t] = ln2_b[t]; sb1[t] = b1[t]; sb2[t] = b2[t]; }
    __syncthreads();

    const int row = blockIdx.x * 256 + t;

    float yc[16];
    #pragma unroll
    for (int j = 0; j < 16; ++j) yc[j] = Yc[(size_t)row * 16 + j];

    // groupnorm over 2 groups of 8
    float gn[16];
    #pragma unroll
    for (int g = 0; g < 2; ++g) {
        float m = 0.f;
        #pragma unroll
        for (int j = 0; j < 8; ++j) m += yc[g * 8 + j];
        m *= 0.125f;
        float v = 0.f;
        #pragma unroll
        for (int j = 0; j < 8; ++j) { float d = yc[g * 8 + j] - m; v += d * d; }
        v *= 0.125f;
        float inv = rsqrtf(v + 1e-5f);
        #pragma unroll
        for (int j = 0; j < 8; ++j) gn[g * 8 + j] = (yc[g * 8 + j] - m) * inv;
    }
    #pragma unroll
    for (int j = 0; j < 16; ++j) gn[j] = gn[j] * sgw[j] + sgb[j];

    float xn[8], xo[8];
    #pragma unroll
    for (int i = 0; i < 8; ++i) { xn[i] = Xn[(size_t)row * 8 + i]; xo[i] = X[(size_t)row * 8 + i]; }

    // gate = swish(Xn @ Wg), msr = (gate*gn) @ Wo, Y = msr + X
    float gg[16];
    #pragma unroll
    for (int j = 0; j < 16; ++j) {
        float s = 0.f;
        #pragma unroll
        for (int i = 0; i < 8; ++i) s += xn[i] * sWg[i * 16 + j];
        gg[j] = (s / (1.f + __expf(-s))) * gn[j];
    }
    float y[8];
    #pragma unroll
    for (int o = 0; o < 8; ++o) {
        float s = 0.f;
        #pragma unroll
        for (int j = 0; j < 16; ++j) s += gg[j] * sWo[j * 8 + o];
        y[o] = s + xo[o];
    }

    // LN2
    float m = 0.f;
    #pragma unroll
    for (int i = 0; i < 8; ++i) m += y[i];
    m *= 0.125f;
    float var = 0.f;
    #pragma unroll
    for (int i = 0; i < 8; ++i) { float d = y[i] - m; var += d * d; }
    var *= 0.125f;
    float inv = rsqrtf(var + 1e-5f);
    float z[8];
    #pragma unroll
    for (int i = 0; i < 8; ++i) z[i] = (y[i] - m) * inv * sl2s[i] + sl2b[i];

    // FFN: gelu(Z@w1+b1)@w2+b2 + Y
    float f1[8];
    #pragma unroll
    for (int o = 0; o < 8; ++o) {
        float a = sb1[o];
        #pragma unroll
        for (int i = 0; i < 8; ++i) a += z[i] * sw1[i * 8 + o];
        f1[o] = 0.5f * a * (1.f + erff(a * 0.70710678118f));
    }
    #pragma unroll
    for (int o = 0; o < 8; ++o) {
        float a = sb2[o];
        #pragma unroll
        for (int i = 0; i < 8; ++i) a += f1[i] * sw2[i * 8 + o];
        Xout[(size_t)row * 8 + o] = a + y[o];
    }
}

// ---------------------------------------------------------------------------
// Kernel 6: fuse li1@li2 (no nonlinearity between) -> Wf (2048x2), bf (2)
// ---------------------------------------------------------------------------
__global__ __launch_bounds__(256) void fuse_li_kernel(
    const float* __restrict__ li1_w, const float* __restrict__ li1_b,
    const float* __restrict__ li2_w, const float* __restrict__ li2_b,
    float* __restrict__ Wf, float* __restrict__ bf)
{
    int k = blockIdx.x * 256 + threadIdx.x;   // 0..2047
    const float* r = li1_w + (size_t)k * 1000;
    float s0 = 0.f, s1 = 0.f;
    for (int j = 0; j < 1000; ++j) {
        float w = r[j];
        s0 += w * li2_w[j * 2 + 0];
        s1 += w * li2_w[j * 2 + 1];
    }
    Wf[k * 2 + 0] = s0;
    Wf[k * 2 + 1] = s1;
    if (k < 2) {
        float s = li2_b[k];
        for (int j = 0; j < 1000; ++j) s += li1_b[j] * li2_w[j * 2 + k];
        bf[k] = s;
    }
}

// ---------------------------------------------------------------------------
// Kernel 7: final GEMV: out (128,2) = Xout (128,2048) @ Wf + bf
// ---------------------------------------------------------------------------
__global__ __launch_bounds__(256) void final_kernel(
    const float* __restrict__ Xout, const float* __restrict__ Wf,
    const float* __restrict__ bf, float* __restrict__ out)
{
    int i = threadIdx.x;            // 0..255
    int b = i >> 1, c = i & 1;
    const float* xr = Xout + (size_t)b * 2048;
    float s = bf[c];
    for (int k = 0; k < 2048; ++k) s += xr[k] * Wf[k * 2 + c];
    out[b * 2 + c] = s;
}

// ---------------------------------------------------------------------------
extern "C" void kernel_launch(void* const* d_in, const int* in_sizes, int n_in,
                              void* d_out, int out_size, void* d_ws, size_t ws_size,
                              hipStream_t stream) {
    (void)in_sizes; (void)n_in; (void)out_size; (void)ws_size;

    const float* x     = (const float*)d_in[0];
    const float* fc_w  = (const float*)d_in[1];
    const float* fc_b  = (const float*)d_in[2];
    const float* ln1_s = (const float*)d_in[3];
    const float* ln1_b = (const float*)d_in[4];
    const float* ln2_s = (const float*)d_in[5];
    const float* ln2_b = (const float*)d_in[6];
    const float* Wq    = (const float*)d_in[7];
    const float* Wk    = (const float*)d_in[8];
    const float* Wv    = (const float*)d_in[9];
    const float* Wg    = (const float*)d_in[10];
    const float* Wo    = (const float*)d_in[11];
    const float* gn_w  = (const float*)d_in[12];
    const float* gn_b  = (const float*)d_in[13];
    const float* ffw1  = (const float*)d_in[14];
    const float* ffb1  = (const float*)d_in[15];
    const float* ffw2  = (const float*)d_in[16];
    const float* ffb2  = (const float*)d_in[17];
    const float* li1_w = (const float*)d_in[18];
    const float* li1_b = (const float*)d_in[19];
    const float* li2_w = (const float*)d_in[20];
    const float* li2_b = (const float*)d_in[21];
    float* out = (float*)d_out;

    // workspace layout: Apk (f16) first, then f32 buffers
    _Float16* Apk  = (_Float16*)d_ws;                             // 128*49152 f16
    float* fbase   = (float*)d_ws + ((size_t)MDIM * KTOT) / 2;    // = +3,145,728 f32
    float* partial = fbase;                                       // 8*128*2048
    float* X       = partial + (size_t)KSPLIT * MDIM * NDIM;      // 128*2048
    float* Xn      = X  + (size_t)MDIM * NDIM;                    // 128*2048
    float* Yc      = Xn + (size_t)MDIM * NDIM;                    // 128*256*16
    float* Xout    = Yc + (size_t)MDIM * 256 * 16;                // 128*2048
    float* Wf      = Xout + (size_t)MDIM * NDIM;                  // 2048*2
    float* bf      = Wf + 4096;                                   // 2

    pack_a_kernel<<<(MDIM * 6144) / 256, 256, 0, stream>>>(x, Apk);
    fc_gemm_kernel<<<dim3(NDIM / TILE_N, KSPLIT), 256, 0, stream>>>(Apk, fc_w, partial);
    fc_reduce_kernel<<<(MDIM * NDIM) / 256, 256, 0, stream>>>(partial, fc_b, X);
    ln1_kernel<<<(MDIM * 256) / 256, 256, 0, stream>>>(X, ln1_s, ln1_b, Xn);
    retention_kernel<<<dim3(MDIM, 2), 256, 0, stream>>>(Xn, Wq, Wk, Wv, Yc);
    post_kernel<<<(MDIM * 256) / 256, 256, 0, stream>>>(
        X, Xn, Yc, Wg, Wo, gn_w, gn_b, ln2_s, ln2_b, ffw1, ffb1, ffw2, ffb2, Xout);
    fuse_li_kernel<<<NDIM / 256, 256, 0, stream>>>(li1_w, li1_b, li2_w, li2_b, Wf, bf);
    final_kernel<<<1, 256, 0, stream>>>(Xout, Wf, bf, out);
}
